// QuadConvLayer_5557687681097
// MI455X (gfx1250) — matrix operand order repaired and checked by reading, not compiled, not run on hardware
//
#include <hip/hip_runtime.h>
#include <hip/hip_bf16.h>

// ---------------------------------------------------------------------------
// QuadConv layer for MI455X (gfx1250), wave32 + WMMA.
//
// Stage 1: build Bt[n][k] = kern[co,ci,m,k] (f16), n = co*144+m, k = ci*576+kq.
//          Bump-gated: ~98% of entries are exactly 0 -> skip the sin-MLP.
// Stage 2: convert features (32 x 9216) to f16.
// Stage 3: GEMM out[32 x 2304] = A[32 x 9216] * B[9216 x 2304] via
//          v_wmma_f32_16x16x32_f16, K split 8 ways into f32 partials (ws).
// Stage 4: deterministic reduction of the 8 partials into d_out.
// ---------------------------------------------------------------------------

#define NQ      24
#define KQ      576                 // 24*24 quadrature points
#define M_OUT   144                 // 12*12 output locations
#define C_IN    16
#define C_OUT   16
#define BATCH   32
#define KG      (C_IN * KQ)         // 9216  GEMM K
#define NG      (C_OUT * M_OUT)     // 2304  GEMM N
#define DECAYC  1296.0f             // (24/4)^4
#define KSPLIT  8
#define KSEG    (KG / KSPLIT)       // 1152 -> 36 WMMA steps per wave
#define OUT_EL  (BATCH * NG)        // 73728

typedef __attribute__((ext_vector_type(16))) _Float16     v16h;
typedef __attribute__((ext_vector_type(8)))  float        v8f;
typedef __attribute__((ext_vector_type(4)))  unsigned int v4u;

union FragU { v16h v; v4u q[2]; };

// ---------------------------------------------------------------------------
// Stage 1: kernel-matrix builder (bump-gated sin-MLP), writes f16 Bt[n*KG + k]
// Thread t: k = t % KG (fastest -> coalesced f16 stores), n = t / KG.
// ---------------------------------------------------------------------------
__global__ void __launch_bounds__(256)
qc_build_bt(const float* __restrict__ locs,   // (144,2)
            const float* __restrict__ qn,     // (24)
            const float* __restrict__ qw,     // (24)
            const float* __restrict__ W1,     // (16,16,2,8)
            const float* __restrict__ W2,     // (16,16,8,8)
            const float* __restrict__ W3,     // (16,16,8,1)
            _Float16* __restrict__ Bt)        // (NG, KG)
{
    const long long t = (long long)blockIdx.x * 256 + threadIdx.x;
    if (t >= (long long)NG * KG) return;

    const int kk = (int)(t % KG);
    const int nn = (int)(t / KG);
    const int ci = kk / KQ, k = kk - ci * KQ;
    const int co = nn / M_OUT, m = nn - co * M_OUT;
    const int iq = k / NQ, jq = k - iq * NQ;

    const float xx = locs[2 * m]     - qn[jq];
    const float xy = locs[2 * m + 1] - qn[iq];
    const float r2 = xx * xx + xy * xy;
    const float da = DECAYC * r2 * r2;

    float val = 0.0f;
    if (da < 1.0f) {
        // bump = e * exp(-1/(1-da))
        const float bump = 2.71828182845904523f * expf(-1.0f / (1.0f - da));

        const int pair = co * C_IN + ci;
        const float* __restrict__ w1 = W1 + pair * 16;   // [2][8]
        const float* __restrict__ w2 = W2 + pair * 64;   // [8][8]
        const float* __restrict__ w3 = W3 + pair * 8;    // [8]

        float h1[8];
        #pragma unroll
        for (int h = 0; h < 8; ++h)
            h1[h] = sinf(xx * w1[h] + xy * w1[8 + h]);

        float w = 0.0f;
        #pragma unroll
        for (int g = 0; g < 8; ++g) {
            float acc = 0.0f;
            #pragma unroll
            for (int h = 0; h < 8; ++h)
                acc = fmaf(h1[h], w2[h * 8 + g], acc);
            w = fmaf(sinf(acc), w3[g], w);
        }
        val = w * bump * (qw[iq] * qw[jq]);
    }
    Bt[t] = (_Float16)val;
}

// ---------------------------------------------------------------------------
// Stage 2: features f32 -> f16 (layout already matches A[b][kk])
// ---------------------------------------------------------------------------
__global__ void __launch_bounds__(256)
qc_cvt_feat(const float* __restrict__ f, _Float16* __restrict__ Ah, int n)
{
    const int i = blockIdx.x * 256 + threadIdx.x;
    if (i < n) Ah[i] = (_Float16)f[i];
}

// ---------------------------------------------------------------------------
// Stage 3: WMMA GEMM, one 16x16 C tile per wave per K-segment.
// Fragment packing follows CDNA5 ISA §7.12.2 (f16 A 16x32, B 32x16,
// f32 C 16x16); both A and B fragments are two 16B-aligned b128 loads per lane.
// ---------------------------------------------------------------------------
__global__ void __launch_bounds__(256)
qc_wmma_gemm(const _Float16* __restrict__ A,    // (32, KG)  row-major
             const _Float16* __restrict__ Bt,   // (NG, KG)  == B^T
             float* __restrict__ partial)       // (KSPLIT, 32, NG)
{
    const int lane = threadIdx.x & 31;
    const int wid  = blockIdx.x * (blockDim.x >> 5) + (threadIdx.x >> 5);

    const int seg    = wid / 288;        // K segment [0,8)
    const int rem    = wid % 288;
    const int base_n = (rem >> 1) << 4;  // 144 N tiles
    const int base_m = (rem & 1) << 4;   // 2  M tiles (batch 32)

    const int r   = lane & 15;           // A row / B,C column within tile
    const int hlf = lane >> 4;

    const _Float16* __restrict__ arow = A  + (size_t)(base_m + r) * KG;
    const _Float16* __restrict__ bcol = Bt + (size_t)(base_n + r) * KG;
    const int k0 = seg * KSEG;

    v8f acc = {};
    #pragma unroll 2
    for (int kb = k0; kb < k0 + KSEG; kb += 32) {
        FragU a, b;
        // A 16x32 f16: lanes<16 hold K {0..7,16..23}, lanes>=16 {8..15,24..31}
        a.q[0] = *reinterpret_cast<const v4u*>(arow + kb + hlf * 8);
        a.q[1] = *reinterpret_cast<const v4u*>(arow + kb + 16 + hlf * 8);
        // B 32x16 f16: lanes<16 hold K {0..7,16..23}, lanes>=16 {8..15,24..31}
        b.q[0] = *reinterpret_cast<const v4u*>(bcol + kb + hlf * 8);
        b.q[1] = *reinterpret_cast<const v4u*>(bcol + kb + 16 + hlf * 8);
        // gfx1250 global_prefetch_b8 for next K step of this lane's B column
        __builtin_prefetch(bcol + kb + 32, 0, 1);

        acc = __builtin_amdgcn_wmma_f32_16x16x32_f16(
                  false, a.v, false, b.v, (short)0, acc, false, false);
    }

    // C layout: VGPR i -> row (i + 8*hlf), col = lane&15
    float* __restrict__ pout = partial + (size_t)seg * OUT_EL;
    #pragma unroll
    for (int i = 0; i < 8; ++i) {
        const int row = base_m + i + hlf * 8;
        pout[(size_t)row * NG + base_n + r] = acc[i];
    }
}

// ---------------------------------------------------------------------------
// Stage 4: fixed-order reduction of K-split partials (deterministic)
// ---------------------------------------------------------------------------
__global__ void __launch_bounds__(256)
qc_reduce(const float* __restrict__ partial, float* __restrict__ out)
{
    const int i = blockIdx.x * 256 + threadIdx.x;
    if (i >= OUT_EL) return;
    float s = 0.0f;
    #pragma unroll
    for (int p = 0; p < KSPLIT; ++p)
        s += partial[(size_t)p * OUT_EL + i];
    out[i] = s;
}

// ---------------------------------------------------------------------------
extern "C" void kernel_launch(void* const* d_in, const int* in_sizes, int n_in,
                              void* d_out, int out_size, void* d_ws, size_t ws_size,
                              hipStream_t stream) {
    const float* features = (const float*)d_in[0];  // (32,16,576)
    const float* locs     = (const float*)d_in[1];  // (144,2)
    const float* qn       = (const float*)d_in[2];  // (24)
    const float* qw       = (const float*)d_in[3];  // (24)
    const float* W1       = (const float*)d_in[4];  // (16,16,2,8)
    const float* W2       = (const float*)d_in[5];  // (16,16,8,8)
    const float* W3       = (const float*)d_in[6];  // (16,16,8,1)
    float*       out      = (float*)d_out;          // (32,16,144)

    // Workspace carve-up (~43.3 MB total)
    char* ws = (char*)d_ws;
    _Float16* Bt      = (_Float16*)ws;                                   // NG*KG f16
    _Float16* Ah      = (_Float16*)(ws + (size_t)NG * KG * 2);           // 32*KG f16
    float*    partial = (float*)   (ws + (size_t)NG * KG * 2
                                       + (size_t)BATCH * KG * 2);        // 8*OUT_EL f32

    // Stage 2 (independent of stage 1; both precede the GEMM in stream order)
    qc_cvt_feat<<<(BATCH * KG + 255) / 256, 256, 0, stream>>>(
        features, Ah, BATCH * KG);

    // Stage 1: 2304*9216 = 21.2M entries, 256/block
    qc_build_bt<<<(unsigned)(((long long)NG * KG) / 256), 256, 0, stream>>>(
        locs, qn, qw, W1, W2, W3, Bt);

    // Stage 3: 2304 waves = 288 blocks * 8 waves
    qc_wmma_gemm<<<288, 256, 0, stream>>>(Ah, Bt, partial);

    // Stage 4
    qc_reduce<<<(OUT_EL + 255) / 256, 256, 0, stream>>>(partial, out);
}
